// GNNBlocks_45827301048730
// MI455X (gfx1250) — matrix operand order repaired and checked by reading, not compiled, not run on hardware
//
#include <hip/hip_runtime.h>
#include <hip/hip_bf16.h>
#include <math.h>

// ---------------------------------------------------------------------------
// CDNA5 / gfx1250 GNN forward (GCN stack + 3x [GAT,GAT,GCN+GLU,LN,GRU-gate])
// Dense GEMMs run on v_wmma_f32_16x16x32_bf16 (wave32, 16x16 tiles, f32 acc),
// K stepped by 64 with two WMMAs per LDS stage. Edge aggregation uses global
// f32 atomics (N*256 tables are L2-resident on the 192MB L2).
// Input order assumes insertion-order flatten of setup_inputs (56 leaves).
// ---------------------------------------------------------------------------

typedef __bf16 bf16_t;
typedef bf16_t v16bf __attribute__((ext_vector_type(16)));
typedef float  v8f   __attribute__((ext_vector_type(8)));

#define GHEADS 4
#define GDHEAD 64

// ------------------------------- utilities --------------------------------

__global__ void k_fill(float* __restrict__ p, long long n, float v) {
  long long i = (long long)blockIdx.x * blockDim.x + threadIdx.x;
  if (i < n) p[i] = v;
}

__device__ inline void atomicMaxF(float* addr, float v) {
  unsigned int* ua = (unsigned int*)addr;
  unsigned int old = __float_as_uint(*addr);
  while (__uint_as_float(old) < v) {
    unsigned int assumed = old;
    old = atomicCAS(ua, assumed, __float_as_uint(v));
    if (old == assumed) break;
  }
}

// pack two f32 -> packed bf16 pair (lowers to v_cvt_pk_bf16_f32)
__device__ inline unsigned pk_bf16(float a, float b) {
  unsigned short la = __builtin_bit_cast(unsigned short, (bf16_t)a);
  unsigned short lb = __builtin_bit_cast(unsigned short, (bf16_t)b);
  return (unsigned)la | ((unsigned)lb << 16);
}

// ------------------------------ degree kernels ----------------------------

__global__ void k_deg(const int* __restrict__ src, const int* __restrict__ dst,
                      float* __restrict__ dsrc, float* __restrict__ ddst, int E) {
  int e = blockIdx.x * blockDim.x + threadIdx.x;
  if (e < E) {
    atomicAdd(&dsrc[src[e]], 1.0f);
    atomicAdd(&ddst[dst[e]], 1.0f);
  }
}

__global__ void k_rsqrt_clamp(float* __restrict__ d, int N) {
  int i = blockIdx.x * blockDim.x + threadIdx.x;
  if (i < N) d[i] = rsqrtf(fmaxf(d[i], 1.0f));
}

// ---------------------- GCN neighbor aggregation (scatter) ----------------
// m[dst] += x[src] * doi[src].  fsh = log2(F/4) so e/f4 come from shift/mask.

__global__ void k_agg(const int* __restrict__ src, const int* __restrict__ dst,
                      const float* __restrict__ doi, const float* __restrict__ x,
                      float* __restrict__ m, int E, int F, int fsh) {
  long long tid = (long long)blockIdx.x * blockDim.x + threadIdx.x;
  long long total = (long long)E << fsh;
  if (tid >= total) return;
  int e  = (int)(tid >> fsh);
  int f4 = ((int)tid & ((1 << fsh) - 1)) << 2;
  int s = src[e], d = dst[e];
  float sc = doi[s];
  const float4 v = *(const float4*)&x[(size_t)s * F + f4];
  float* mp = &m[(size_t)d * F + f4];
  atomicAdd(mp + 0, v.x * sc);
  atomicAdd(mp + 1, v.y * sc);
  atomicAdd(mp + 2, v.z * sc);
  atomicAdd(mp + 3, v.w * sc);
}

// ------------------------- WMMA bf16 GEMM (fused) -------------------------
// C[M x Fo] = act( A[M x K] @ W[K x Fo] (+ A2[M x K2] @ W2) + bias ),
// RS: fuse per-row scale on A (D_in^-1/2). Block = 256 thr = 8 waves.
// Tile: 64(M) x 32(N), each wave one 16x16 WMMA subtile, K stepped by 64
// (two v_wmma per LDS stage). OOB rows are CLAMPED during staging (their
// products only feed C rows we never store), so staging is branch-free.
// Epilogue: wave-uniform subtile guard (M%16==0 fast path) + strength-
// reduced store addressing. ACT: 0=none 1=relu 2=sigmoid 3=tanh

template <int ACT, bool RS>
__global__ __launch_bounds__(256)
void k_gemm(const float* __restrict__ A, const float* __restrict__ W,
            const float* __restrict__ A2, const float* __restrict__ W2,
            const float* __restrict__ bias, const float* __restrict__ rowScale,
            float* __restrict__ C, int M, int K, int K2, int Fo) {
  __shared__ bf16_t As[64][68];   // [row][k], stride 68 elems = 136B (8B-aligned pairs)
  __shared__ bf16_t Bs[32][68];   // [n][k]   transposed so (k,k+1) pairs contiguous

  const int t = threadIdx.x;
  const int rowBase = blockIdx.x * 64;
  const int colBase = blockIdx.y * 32;
  const int lane = t & 31;
  const int wave = t >> 5;
  const int subM = wave >> 1;   // 0..3
  const int subN = wave & 1;    // 0..1
  const int nloc = lane & 15;   // N index (and C/D column) within 16
  const int g    = lane >> 4;   // half-wave select

  v8f acc;
  const float bv0 = (bias != nullptr) ? bias[colBase + subN * 16 + nloc] : 0.0f;
#pragma unroll
  for (int r = 0; r < 8; ++r) acc[r] = bv0;

  const int npass = (A2 != nullptr) ? 2 : 1;
  for (int pass = 0; pass < npass; ++pass) {
    const float* Ap = pass ? A2 : A;
    const float* Wp = pass ? W2 : W;
    const int Kp = pass ? K2 : K;
    for (int k0 = 0; k0 < Kp; k0 += 64) {
      __syncthreads();
      // Stage A tile 64 rows x 64 k: 4 x float4 per thread, packed bf16 stores.
#pragma unroll
      for (int i = 0; i < 4; ++i) {
        int idx = t + 256 * i;
        int r  = idx >> 4;            // 0..63
        int c4 = (idx & 15) << 2;     // 0..60 step 4
        int grow = rowBase + r;
        grow = (grow < M) ? grow : (M - 1);   // clamp: branch-free staging
        const float4 va = *(const float4*)&Ap[(size_t)grow * Kp + (k0 + c4)];
        float sc = 1.0f;
        if (RS) sc = rowScale[grow];
        uint2 p;
        p.x = pk_bf16(va.x * sc, va.y * sc);
        p.y = pk_bf16(va.z * sc, va.w * sc);
        *(uint2*)&As[r][c4] = p;
      }
      // Stage B tile 64 k x 32 n transposed: 2 x float4 per thread.
#pragma unroll
      for (int i = 0; i < 2; ++i) {
        int idx = t + 256 * i;
        int kk = idx >> 3;            // 0..63
        int n4 = (idx & 7) << 2;      // 0..28 step 4
        const float4 vb = *(const float4*)&Wp[(size_t)(k0 + kk) * Fo + (colBase + n4)];
        Bs[n4 + 0][kk] = (bf16_t)vb.x;
        Bs[n4 + 1][kk] = (bf16_t)vb.y;
        Bs[n4 + 2][kk] = (bf16_t)vb.z;
        Bs[n4 + 3][kk] = (bf16_t)vb.w;
      }
      __syncthreads();

      // Two K=32 WMMAs per stage; per-lane operand gather per CDNA5 16-bit
      // WMMA VGPR layouts (A and B: same 8-wide interleave over k).
#pragma unroll
      for (int kh = 0; kh < 2; ++kh) {
        const int kb0 = kh * 32;
        v16bf av, bvv;
#pragma unroll
        for (int j = 0; j < 8; ++j) {
          int ka = kb0 + ((j < 4) ? (2 * j) : (16 + 2 * (j - 4))) + 8 * g;
          av[2 * j]     = As[subM * 16 + nloc][ka];
          av[2 * j + 1] = As[subM * 16 + nloc][ka + 1];
          int kb = kb0 + ((j < 4) ? (2 * j) : (16 + 2 * (j - 4))) + 8 * g;
          bvv[2 * j]     = Bs[subN * 16 + nloc][kb];
          bvv[2 * j + 1] = Bs[subN * 16 + nloc][kb + 1];
        }
        acc = __builtin_amdgcn_wmma_f32_16x16x32_bf16(false, av, false, bvv,
                                                      (short)0, acc, false, false);
      }
    }
  }

  // C/D layout: acc[r] -> C[row = r + 8*g][col = nloc] of the 16x16 subtile.
  float vals[8];
#pragma unroll
  for (int r = 0; r < 8; ++r) {
    float v = acc[r];
    if (ACT == 1)      v = fmaxf(v, 0.0f);
    else if (ACT == 2) v = 1.0f / (1.0f + __expf(-v));
    else if (ACT == 3) v = tanhf(v);
    vals[r] = v;
  }
  const int subRow = rowBase + subM * 16;            // wave-uniform
  float* Cp = C + (size_t)(subRow + 8 * g) * Fo + (colBase + subN * 16 + nloc);
  if (subRow + 16 <= M) {                            // uniform fast path
#pragma unroll
    for (int r = 0; r < 8; ++r) { *Cp = vals[r]; Cp += Fo; }
  } else {
#pragma unroll
    for (int r = 0; r < 8; ++r) {
      if (subRow + 8 * g + r < M) *Cp = vals[r];
      Cp += Fo;
    }
  }
}

// ------------------------------- GAT kernels ------------------------------

__global__ void k_gat_scores(const float* __restrict__ h,
                             const float* __restrict__ al,
                             const float* __restrict__ ar,
                             float* __restrict__ el, float* __restrict__ er,
                             int N) {
  int tid = blockIdx.x * blockDim.x + threadIdx.x;
  if (tid >= N * GHEADS) return;
  int n = tid >> 2, hd = tid & 3;
  const float* hp = &h[(size_t)n * (GHEADS * GDHEAD) + hd * GDHEAD];
  const float* alp = &al[hd * GDHEAD];
  const float* arp = &ar[hd * GDHEAD];
  float sl = 0.0f, sr = 0.0f;
#pragma unroll 8
  for (int d = 0; d < GDHEAD; ++d) { sl += hp[d] * alp[d]; sr += hp[d] * arp[d]; }
  el[tid] = sl;
  er[tid] = sr;
}

__global__ void k_edge_max(const int* __restrict__ src, const int* __restrict__ dst,
                           const float* __restrict__ el, const float* __restrict__ er,
                           float* __restrict__ emax, int E) {
  int tid = blockIdx.x * blockDim.x + threadIdx.x;
  if (tid >= E * GHEADS) return;
  int e = tid >> 2, hd = tid & 3;
  float v = el[src[e] * GHEADS + hd] + er[dst[e] * GHEADS + hd];
  v = (v >= 0.0f) ? v : 0.2f * v;  // leaky_relu 0.2
  atomicMaxF(&emax[dst[e] * GHEADS + hd], v);
}

__global__ void k_edge_ex(const int* __restrict__ src, const int* __restrict__ dst,
                          const float* __restrict__ el, const float* __restrict__ er,
                          const float* __restrict__ emax, float* __restrict__ exb,
                          float* __restrict__ denom, int E) {
  int tid = blockIdx.x * blockDim.x + threadIdx.x;
  if (tid >= E * GHEADS) return;
  int e = tid >> 2, hd = tid & 3;
  int d4 = dst[e] * GHEADS + hd;
  float v = el[src[e] * GHEADS + hd] + er[d4];
  v = (v >= 0.0f) ? v : 0.2f * v;
  float em = emax[d4];
  em = (em > -1e29f) ? em : 0.0f;  // zero in-degree safety (matches reference)
  float ex = __expf(v - em);
  exb[tid] = ex;
  atomicAdd(&denom[d4], ex);
}

__global__ void k_gat_scatter(const int* __restrict__ src, const int* __restrict__ dst,
                              const float* __restrict__ h, const float* __restrict__ exb,
                              const float* __restrict__ denom, float* __restrict__ rst,
                              int E) {
  // thread = (edge, head, 16-wide d-chunk); all index math shift/mask.
  long long tid = (long long)blockIdx.x * blockDim.x + threadIdx.x;
  long long total = (long long)E * GHEADS * (GDHEAD / 16);
  if (tid >= total) return;
  int c  = (int)(tid & 3);
  int hd = (int)((tid >> 2) & 3);
  int e  = (int)(tid >> 4);
  int s = src[e], d = dst[e];
  float alpha = exb[e * GHEADS + hd] / denom[d * GHEADS + hd];
  const float* hp = &h[(size_t)s * (GHEADS * GDHEAD) + hd * GDHEAD + c * 16];
  float* rp = &rst[(size_t)d * (GHEADS * GDHEAD) + hd * GDHEAD + c * 16];
#pragma unroll
  for (int i = 0; i < 16; ++i) atomicAdd(rp + i, hp[i] * alpha);
}

__global__ void k_bias_relu(float* __restrict__ x, const float* __restrict__ b,
                            long long n, int fmask) {
  long long i = (long long)blockIdx.x * blockDim.x + threadIdx.x;
  if (i < n) x[i] = fmaxf(x[i] + b[(int)i & fmask], 0.0f);
}

// ------------------------- GLU + residual + scale -------------------------
// h2 = (h1 + wide[:,:256] * sigmoid(wide[:,256:])) * sqrt(0.5)

__global__ void k_glu(const float* __restrict__ h1, const float* __restrict__ wide,
                      float* __restrict__ h2, long long n) {
  long long i = (long long)blockIdx.x * blockDim.x + threadIdx.x;
  if (i >= n) return;
  long long row = i >> 8;
  int j = (int)(i & 255);
  float a = wide[row * 512 + j];
  float bb = wide[row * 512 + 256 + j];
  float gv = a * (1.0f / (1.0f + __expf(-bb)));
  h2[i] = (h1[i] + gv) * 0.70710678118654752f;
}

// ------------------------------ LayerNorm(256) ----------------------------
// One wave32 per row, 8 floats per lane, shfl_xor reductions.

__global__ __launch_bounds__(256)
void k_layernorm(float* __restrict__ x, const float* __restrict__ g,
                 const float* __restrict__ b, int N) {
  int row = blockIdx.x * 8 + (threadIdx.x >> 5);
  int lane = threadIdx.x & 31;
  if (row >= N) return;
  float v[8];
  float s = 0.0f;
#pragma unroll
  for (int i = 0; i < 8; ++i) { v[i] = x[(size_t)row * 256 + lane + 32 * i]; s += v[i]; }
#pragma unroll
  for (int off = 16; off > 0; off >>= 1) s += __shfl_xor(s, off);
  float mu = s * (1.0f / 256.0f);
  float vs = 0.0f;
#pragma unroll
  for (int i = 0; i < 8; ++i) { float d = v[i] - mu; vs += d * d; }
#pragma unroll
  for (int off = 16; off > 0; off >>= 1) vs += __shfl_xor(vs, off);
  float inv = rsqrtf(vs * (1.0f / 256.0f) + 1e-5f);
#pragma unroll
  for (int i = 0; i < 8; ++i) {
    int j = lane + 32 * i;
    x[(size_t)row * 256 + j] = (v[i] - mu) * inv * g[j] + b[j];
  }
}

// ----------------------------- gate elementwise ---------------------------

__global__ void k_mul_inplace(float* __restrict__ a, const float* __restrict__ b,
                              long long n) {
  long long i = (long long)blockIdx.x * blockDim.x + threadIdx.x;
  if (i < n) a[i] = a[i] * b[i];
}

__global__ void k_gate_update(float* __restrict__ h1, const float* __restrict__ z,
                              const float* __restrict__ h2n, long long n) {
  long long i = (long long)blockIdx.x * blockDim.x + threadIdx.x;
  if (i < n) h1[i] = z[i] * h1[i] + (1.0f - z[i]) * h2n[i];
}

// --------------------------------- pooling --------------------------------

__global__ void k_pool(const int* __restrict__ gid, const float* __restrict__ h,
                       float* __restrict__ out, float* __restrict__ cnt,
                       long long total, int fsh, int fmask) {
  long long i = (long long)blockIdx.x * blockDim.x + threadIdx.x;
  if (i >= total) return;
  int n = (int)(i >> fsh);
  int j = (int)i & fmask;
  int gd = gid[n];
  atomicAdd(&out[((size_t)gd << fsh) + j], h[i]);
  if (j == 0) atomicAdd(&cnt[gd], 1.0f);
}

__global__ void k_pool_div(float* __restrict__ out, const float* __restrict__ cnt,
                           int total, int fsh) {
  int i = blockIdx.x * blockDim.x + threadIdx.x;
  if (i < total) out[i] /= fmaxf(cnt[i >> fsh], 1.0f);
}

// --------------------------------- driver ---------------------------------

static inline long long cdivll(long long a, long long b) { return (a + b - 1) / b; }
static inline int ilog2(int x) { int s = 0; while ((1 << s) < x) ++s; return s; }

extern "C" void kernel_launch(void* const* d_in, const int* in_sizes, int n_in,
                              void* d_out, int out_size, void* d_ws, size_t ws_size,
                              hipStream_t stream) {
  (void)n_in; (void)ws_size;
  // ---- input mapping (insertion-order flatten of setup_inputs) ----
  const int*   src   = (const int*)  d_in[0];
  const int*   dst   = (const int*)  d_in[1];
  const int*   gid   = (const int*)  d_in[2];
  const float* feats = (const float*)d_in[3];
  // d_in[4] = n_graphs scalar (device); we use out_size instead.
  const int E = in_sizes[0];
  const int N = in_sizes[2];
  const int IN  = in_sizes[3] / N;       // 64
  const int HID = in_sizes[6];           // 128 (gcn0.layers[0].b)
  const int W2  = in_sizes[10];          // 256 (gcn0.out.b)
  const int W4  = in_sizes[20];          // 512 (blocks[0].gcn.b)
  const int NG  = out_size / W2;

  const float* g0W0 = (const float*)d_in[5];
  const float* g0b0 = (const float*)d_in[6];
  const float* g0W1 = (const float*)d_in[7];
  const float* g0b1 = (const float*)d_in[8];
  const float* g0Wo = (const float*)d_in[9];
  const float* g0bo = (const float*)d_in[10];
  // blocks: base 11 + blk*10 ; lns: base 41 + blk*2 ; gate: base 47
  const float* wr1 = (const float*)d_in[47];
  const float* wr2 = (const float*)d_in[48];
  const float* br  = (const float*)d_in[49];
  const float* wz1 = (const float*)d_in[50];
  const float* wz2 = (const float*)d_in[51];
  const float* bz  = (const float*)d_in[52];
  const float* wh1 = (const float*)d_in[53];
  const float* wh2 = (const float*)d_in[54];
  const float* bh  = (const float*)d_in[55];

  // ---- workspace carve-up (floats) ----
  float* ws = (float*)d_ws;
  size_t off = 0;
  auto carve = [&](size_t nf) { float* p = ws + off; off += nf; return p; };
  float* H1   = carve((size_t)N * W2);
  float* G1   = carve((size_t)N * W2);
  float* G2   = carve((size_t)N * W2);
  float* H2   = carve((size_t)N * W2);
  float* Rb   = carve((size_t)N * W2);
  float* Zb   = carve((size_t)N * W2);
  float* wide = carve((size_t)N * W4);
  float* doi  = carve((size_t)N);
  float* dii  = carve((size_t)N);
  float* el   = carve((size_t)N * GHEADS);
  float* er   = carve((size_t)N * GHEADS);
  float* emax = carve((size_t)N * GHEADS);
  float* denom= carve((size_t)N * GHEADS);
  float* exb  = carve((size_t)E * GHEADS);
  float* cnt  = carve((size_t)NG);

  const int TB = 256;
  auto fill = [&](float* p, long long n, float v) {
    k_fill<<<(unsigned)cdivll(n, TB), TB, 0, stream>>>(p, n, v);
  };
  // act: 0=none 1=relu(+rowScale) 2=sigmoid 3=tanh
  auto gemm = [&](const float* A, const float* W, const float* A2, const float* W2p,
                  const float* bias, const float* rs, float* C,
                  int M, int K, int K2, int Fo, int act) {
    dim3 grid((unsigned)cdivll(M, 64), (unsigned)(Fo / 32));
    dim3 blk(256);
    switch (act) {
      case 0: k_gemm<0, false><<<grid, blk, 0, stream>>>(A, W, A2, W2p, bias, rs, C, M, K, K2, Fo); break;
      case 1: k_gemm<1, true ><<<grid, blk, 0, stream>>>(A, W, A2, W2p, bias, rs, C, M, K, K2, Fo); break;
      case 2: k_gemm<2, false><<<grid, blk, 0, stream>>>(A, W, A2, W2p, bias, rs, C, M, K, K2, Fo); break;
      default:k_gemm<3, false><<<grid, blk, 0, stream>>>(A, W, A2, W2p, bias, rs, C, M, K, K2, Fo); break;
    }
  };
  auto agg = [&](const float* x, float* m, int F) {
    fill(m, (long long)N * F, 0.0f);
    int fsh = ilog2(F / 4);
    long long tot = (long long)E << fsh;
    k_agg<<<(unsigned)cdivll(tot, TB), TB, 0, stream>>>(src, dst, doi, x, m, E, F, fsh);
  };
  auto gat = [&](const float* x, float* out, const float* gW, const float* gal,
                 const float* gar, const float* gb) {
    gemm(x, gW, nullptr, nullptr, nullptr, nullptr, wide, N, W2, 0, W2, 0);
    k_gat_scores<<<(unsigned)cdivll((long long)N * GHEADS, TB), TB, 0, stream>>>(
        wide, gal, gar, el, er, N);
    fill(emax, (long long)N * GHEADS, -1e30f);
    fill(denom, (long long)N * GHEADS, 0.0f);
    k_edge_max<<<(unsigned)cdivll((long long)E * GHEADS, TB), TB, 0, stream>>>(
        src, dst, el, er, emax, E);
    k_edge_ex<<<(unsigned)cdivll((long long)E * GHEADS, TB), TB, 0, stream>>>(
        src, dst, el, er, emax, exb, denom, E);
    fill(out, (long long)N * W2, 0.0f);
    long long stot = (long long)E * GHEADS * (GDHEAD / 16);
    k_gat_scatter<<<(unsigned)cdivll(stot, TB), TB, 0, stream>>>(
        src, dst, wide, exb, denom, out, E);
    k_bias_relu<<<(unsigned)cdivll((long long)N * W2, TB), TB, 0, stream>>>(
        out, gb, (long long)N * W2, W2 - 1);
  };

  // ---- degrees: D_out^-1/2 (src), D_in^-1/2 (dst) ----
  fill(doi, N, 0.0f);
  fill(dii, N, 0.0f);
  k_deg<<<(unsigned)cdivll(E, TB), TB, 0, stream>>>(src, dst, doi, dii, E);
  k_rsqrt_clamp<<<(unsigned)cdivll(N, TB), TB, 0, stream>>>(doi, N);
  k_rsqrt_clamp<<<(unsigned)cdivll(N, TB), TB, 0, stream>>>(dii, N);

  // ---- GCN block: 64 -> 128 -> 128 -> 256 (agg then GEMM+bias+relu) ----
  agg(feats, G1, IN);
  gemm(G1, g0W0, nullptr, nullptr, g0b0, dii, G2, N, IN, 0, HID, 1);
  agg(G2, G1, HID);
  gemm(G1, g0W1, nullptr, nullptr, g0b1, dii, H2, N, HID, 0, HID, 1);
  agg(H2, G1, HID);
  gemm(G1, g0Wo, nullptr, nullptr, g0bo, dii, H1, N, HID, 0, W2, 1);

  // ---- 3 blocks ----
  for (int blk = 0; blk < 3; ++blk) {
    const int bb = 11 + blk * 10;
    const float* gat0W  = (const float*)d_in[bb + 0];
    const float* gat0al = (const float*)d_in[bb + 1];
    const float* gat0ar = (const float*)d_in[bb + 2];
    const float* gat0b  = (const float*)d_in[bb + 3];
    const float* gat1W  = (const float*)d_in[bb + 4];
    const float* gat1al = (const float*)d_in[bb + 5];
    const float* gat1ar = (const float*)d_in[bb + 6];
    const float* gat1b  = (const float*)d_in[bb + 7];
    const float* bgW    = (const float*)d_in[bb + 8];
    const float* bgB    = (const float*)d_in[bb + 9];
    const float* lng    = (const float*)d_in[41 + blk * 2];
    const float* lnb    = (const float*)d_in[42 + blk * 2];

    gat(H1, G1, gat0W, gat0al, gat0ar, gat0b);   // h1 = GAT(h_1)   -> G1
    gat(G1, G2, gat1W, gat1al, gat1ar, gat1b);   // h1 = GAT(h1)    -> G2

    agg(G2, G1, W2);                              // GraphConv 256->512
    gemm(G1, bgW, nullptr, nullptr, bgB, dii, wide, N, W2, 0, W4, 1);

    k_glu<<<(unsigned)cdivll((long long)N * W2, TB), TB, 0, stream>>>(
        G2, wide, H2, (long long)N * W2);          // h2 = (h1+GLU)*sqrt(.5)
    k_layernorm<<<(unsigned)cdivll(N, 8), 256, 0, stream>>>(H2, lng, lnb, N);

    // GRU-style gate (two-input fused WMMA GEMMs)
    gemm(H1, wr1, H2, wr2, br, nullptr, Rb, N, W2, W2, W2, 2);  // R = sigmoid
    gemm(H1, wz1, H2, wz2, bz, nullptr, Zb, N, W2, W2, W2, 2);  // Z = sigmoid
    k_mul_inplace<<<(unsigned)cdivll((long long)N * W2, TB), TB, 0, stream>>>(
        Rb, H1, (long long)N * W2);                              // Rb = R*h_1
    gemm(H2, wh1, Rb, wh2, bh, nullptr, G2, N, W2, W2, W2, 3);  // h2n = tanh
    k_gate_update<<<(unsigned)cdivll((long long)N * W2, TB), TB, 0, stream>>>(
        H1, Zb, G2, (long long)N * W2);            // h_1 = Z*h_1+(1-Z)*h2n
  }

  // ---- mean pooling per graph ----
  float* outp = (float*)d_out;
  fill(outp, (long long)NG * W2, 0.0f);
  fill(cnt, NG, 0.0f);
  {
    int fsh = ilog2(W2);
    k_pool<<<(unsigned)cdivll((long long)N * W2, TB), TB, 0, stream>>>(
        gid, H1, outp, cnt, (long long)N * W2, fsh, W2 - 1);
    k_pool_div<<<(unsigned)cdivll((long long)NG * W2, TB), TB, 0, stream>>>(
        outp, cnt, NG * W2, fsh);
  }
}
